// RAT_21620865368805
// MI455X (gfx1250) — matrix operand order repaired and checked by reading, not compiled, not run on hardware
//
#include <hip/hip_runtime.h>
#include <hip/hip_bf16.h>
#include <cstdint>

// ---------------------------------------------------------------------------
// Problem constants (match reference)
// ---------------------------------------------------------------------------
static constexpr int D_MODEL  = 1024;
static constexpr int NUM_HEAD = 16;
static constexpr int CHUNK    = 64;
static constexpr int NGROUPS  = 2;
static constexpr int D_HEAD   = D_MODEL / NUM_HEAD;          // 64
static constexpr int PROJ_DIM = 3 * D_MODEL + 2 * NGROUPS * D_HEAD; // 3328
static constexpr int BS       = 2;
static constexpr int SEQ      = 4096;
static constexpr int ROWS     = BS * SEQ;                    // 8192
static constexpr int NC       = SEQ / CHUNK;                 // 64

typedef __bf16 bf16_t;
typedef __attribute__((ext_vector_type(8)))  __bf16 v8bf;
typedef __attribute__((ext_vector_type(16))) __bf16 v16bf;
typedef __attribute__((ext_vector_type(8)))  float  v8f;

// ---------------------------------------------------------------------------
// Fragment loaders matching CDNA5 WMMA 16-bit VGPR layouts (ISA 7.12.2).
//
// A (16x32, MxK, row-major, ld = K):
//   lanes 0-15 : M = lane,     VGPR0-3 = K 0..7,   VGPR4-7 = K 16..23
//   lanes 16-31: M = lane-16,  VGPR0-3 = K 8..15,  VGPR4-7 = K 24..31
// B (32x16, KxN) consumed from transposed (N x K row-major) storage:
//   lanes 0-15 : N = lane,     VGPR0-3 = K 0..7,   VGPR4-7 = K 16..23
//   lanes 16-31: N = lane-16,  VGPR0-3 = K 8..15,  VGPR4-7 = K 24..31
// ---------------------------------------------------------------------------
static __device__ inline v16bf cat8(v8bf lo, v8bf hi) {
  return __builtin_shufflevector(lo, hi, 0, 1, 2, 3, 4, 5, 6, 7,
                                 8, 9, 10, 11, 12, 13, 14, 15);
}

static __device__ inline v16bf load_a_frag(const bf16_t* __restrict__ p) {
  v8bf lo = *reinterpret_cast<const v8bf*>(p);
  v8bf hi = *reinterpret_cast<const v8bf*>(p + 16);
  return cat8(lo, hi);
}

static __device__ inline v16bf load_b_frag(const bf16_t* __restrict__ p) {
  v8bf lo = *reinterpret_cast<const v8bf*>(p);
  v8bf hi = *reinterpret_cast<const v8bf*>(p + 16);
  return cat8(lo, hi);
}

// A-fragment sourced from LDS / generic address space (emits ds_load_b128)
static __device__ inline v16bf load_a_frag_lds(const bf16_t* p) {
  v8bf lo = *reinterpret_cast<const v8bf*>(p);
  v8bf hi = *reinterpret_cast<const v8bf*>(p + 16);
  return cat8(lo, hi);
}

static __device__ inline v8f wmma_bf16(v16bf a, v16bf b, v8f c) {
  return __builtin_amdgcn_wmma_f32_16x16x32_bf16(false, a, false, b,
                                                 (short)0, c, false, false);
}

// ---------------------------------------------------------------------------
// Kernel: convert + transpose fp32 weights (KxN row-major) -> bf16 (NxK)
// ---------------------------------------------------------------------------
__global__ void wt_convert_kernel(const float* __restrict__ W,
                                  bf16_t* __restrict__ Wt, int K, int N) {
  size_t idx = (size_t)blockIdx.x * blockDim.x + threadIdx.x;
  size_t total = (size_t)K * N;
  if (idx >= total) return;
  int n = (int)(idx % N);
  int k = (int)(idx / N);
  Wt[(size_t)n * K + k] = (bf16_t)W[idx];
}

// ---------------------------------------------------------------------------
// Kernel: RMSNorm -> bf16 activations (one block per row)
// ---------------------------------------------------------------------------
__global__ __launch_bounds__(256) void rmsnorm_kernel(
    const float* __restrict__ x, const float* __restrict__ w,
    bf16_t* __restrict__ out) {
  int row = blockIdx.x;
  const float* xr = x + (size_t)row * D_MODEL;
  float ss = 0.f;
  for (int i = threadIdx.x; i < D_MODEL; i += 256) {
    float v = xr[i];
    ss += v * v;
  }
  #pragma unroll
  for (int off = 16; off > 0; off >>= 1) ss += __shfl_xor(ss, off, 32);
  __shared__ float red[8];
  if ((threadIdx.x & 31) == 0) red[threadIdx.x >> 5] = ss;
  __syncthreads();
  float tot = 0.f;
  #pragma unroll
  for (int i = 0; i < 8; ++i) tot += red[i];
  float inv = rsqrtf(tot * (1.0f / D_MODEL) + 1e-6f);
  for (int i = threadIdx.x; i < D_MODEL; i += 256) {
    out[(size_t)row * D_MODEL + i] = (bf16_t)(xr[i] * inv * w[i]);
  }
}

// ---------------------------------------------------------------------------
// Kernel: bf16 WMMA GEMM.  C[M,N] = A[M,K] * Bt[N,K]^T (+ residual)
// Block tile 128(M) x 64(N), 8 waves, each wave 32x32 (2x2 WMMA tiles).
// ---------------------------------------------------------------------------
__global__ __launch_bounds__(256) void gemm_bf16_kernel(
    const bf16_t* __restrict__ A, const bf16_t* __restrict__ Bt,
    float* __restrict__ C, const float* __restrict__ residual,
    int M, int N, int K) {
  const int tid  = threadIdx.x;
  const int lane = tid & 31;
  const int wid  = tid >> 5;
  const int wm   = wid & 3;
  const int wn   = wid >> 2;
  const int m0   = blockIdx.x * 128 + wm * 32;
  const int n0   = blockIdx.y * 64 + wn * 32;

  const int r16   = lane & 15;
  const int koffA = (lane & 16) ? 8 : 0;
  const int koffB = (lane & 16) ? 8 : 0;

  const bf16_t* a0 = A + (size_t)(m0 + r16) * K + koffA;
  const bf16_t* a1 = a0 + (size_t)16 * K;
  const bf16_t* b0 = Bt + (size_t)(n0 + r16) * K + koffB;
  const bf16_t* b1 = b0 + (size_t)16 * K;

  v8f acc00 = {}, acc01 = {}, acc10 = {}, acc11 = {};

  for (int k0 = 0; k0 < K; k0 += 32) {
    __builtin_prefetch((const void*)(a0 + k0 + 32), 0, 1);
    __builtin_prefetch((const void*)(b0 + k0 + 32), 0, 1);

    v16bf af0 = load_a_frag(a0 + k0);
    v16bf af1 = load_a_frag(a1 + k0);
    v16bf bf0 = load_b_frag(b0 + k0);
    v16bf bf1 = load_b_frag(b1 + k0);

    acc00 = wmma_bf16(af0, bf0, acc00);
    acc01 = wmma_bf16(af0, bf1, acc01);
    acc10 = wmma_bf16(af1, bf0, acc10);
    acc11 = wmma_bf16(af1, bf1, acc11);
  }

  const int rbase = m0 + ((lane & 16) ? 8 : 0);
  const int col0  = n0 + r16;
  #pragma unroll
  for (int r = 0; r < 8; ++r) {
    int rowA = rbase + r;
    int rowB = rbase + 16 + r;
    float v00 = acc00[r], v01 = acc01[r], v10 = acc10[r], v11 = acc11[r];
    if (residual) {
      v00 += residual[(size_t)rowA * N + col0];
      v01 += residual[(size_t)rowA * N + col0 + 16];
      v10 += residual[(size_t)rowB * N + col0];
      v11 += residual[(size_t)rowB * N + col0 + 16];
    }
    C[(size_t)rowA * N + col0]      = v00;
    C[(size_t)rowA * N + col0 + 16] = v01;
    C[(size_t)rowB * N + col0]      = v10;
    C[(size_t)rowB * N + col0 + 16] = v11;
  }
}

// ---------------------------------------------------------------------------
// Kernel: gated in-chunk scan.  One thread per (batch*chunk, channel).
// channels 0..1023 -> x part, 1024..2047 -> k part (k tiled mod 128).
// Chunk summaries (scan value at t==0) are emitted in bf16, in the two
// layouts the WMMA attention kernel consumes:
//   ckb  [bs*NC][D_MODEL]        (chunk-major rows, channel contiguous)  == B for scores
//   cxbT [bs][D_MODEL][NC]       (channel rows, chunk contiguous)        == B for output
// ---------------------------------------------------------------------------
__global__ __launch_bounds__(256) void scan_kernel(
    const float* __restrict__ proj, const float* __restrict__ gate_bias,
    float* __restrict__ intra_x, float* __restrict__ intra_k,
    bf16_t* __restrict__ ckb, bf16_t* __restrict__ cxbT) {
  int idx = blockIdx.x * 256 + threadIdx.x;
  const int total = BS * NC * 2 * D_MODEL;  // 262144
  if (idx >= total) return;
  int ch      = idx & (2 * D_MODEL - 1);
  int chunkId = idx >> 11;                  // b*NC + chunk  (0..127)
  int rowBase = chunkId * CHUNK;            // == b*SEQ + chunk*CHUNK

  bool isK = ch >= D_MODEL;
  int  c   = ch & (D_MODEL - 1);
  int  gcol = 2 * D_MODEL + c;
  int  vcol = isK ? (3 * D_MODEL + NGROUPS * D_HEAD + (c & 127))  // k cols [3200,3328)
                  : (D_MODEL + c);                                // x cols [1024,2048)
  float bias = gate_bias[c];
  float state = 0.f;
  float* dst = isK ? intra_k : intra_x;

  for (int t = 0; t < CHUNK; ++t) {
    size_t r = (size_t)(rowBase + t);
    const float* pr = proj + r * PROJ_DIM;
    float g = 1.f / (1.f + __expf(-(pr[gcol] + bias)));
    float v = (1.f - g) * pr[vcol];
    state = g * state + v;
    dst[r * D_MODEL + c] = state;
    if (t == 0) {
      if (isK) {
        ckb[(size_t)chunkId * D_MODEL + c] = (bf16_t)state;
      } else {
        int b = chunkId >> 6, chunk = chunkId & (NC - 1);
        cxbT[((size_t)b * D_MODEL + c) * NC + chunk] = (bf16_t)state;
      }
    }
  }
}

// ---------------------------------------------------------------------------
// Kernel: chunk-summary attention, fully on the WMMA pipe.
// One block (128 threads / 4 waves) per (batch, head, chunk) = 64 positions.
//   phase 1: scores[64s x 64c] = (q * dh^-0.5) @ chunk_k^T   (bf16 WMMA)
//   phase 2: per-row masked softmax + intra-chunk term       (scalar, LDS)
//   phase 3: out[64s x 64d]   = w @ chunk_x                  (bf16 WMMA,
//            A-fragments from LDS -> ds_load_b128)
//   epilogue: out = (out + iw*intra_x) / denom * sigmoid(z) -> bf16
// ---------------------------------------------------------------------------
__global__ __launch_bounds__(128) void attn_wmma_kernel(
    const float* __restrict__ proj,
    const float* __restrict__ intra_x, const float* __restrict__ intra_k,
    const bf16_t* __restrict__ ckb, const bf16_t* __restrict__ cxbT,
    bf16_t* __restrict__ zout) {
  const int bid = blockIdx.x;
  const int cs  = bid & (NC - 1);               // this block's chunk index
  const int h   = (bid >> 6) & (NUM_HEAD - 1);
  const int b   = bid >> 10;
  const int srow0 = b * SEQ + cs * CHUNK;       // global row of local s=0
  const int qb  = 3 * D_MODEL + ((h & 1) << 6); // q cols contiguous after tiling
  const int hb  = h * D_HEAD;

  const int tid   = threadIdx.x;
  const int lane  = tid & 31;
  const int wid   = tid >> 5;                   // 0..3 -> M tile
  const int m0    = wid * 16;
  const int r16   = lane & 15;
  const bool hi   = (lane & 16) != 0;
  const int koffA = hi ? 8 : 0;
  const int koffB = hi ? 8 : 0;

  __shared__ __align__(16) float  s_sc[CHUNK * NC];   // scores [s][c], 16 KB
  __shared__ __align__(16) bf16_t s_w[CHUNK * NC];    // softmax weights, 8 KB
  __shared__ float s_iw[CHUNK];
  __shared__ float s_rd[CHUNK];

  const float scale = 0.125f;                   // d_head^-0.5

  // ---- phase 1: scores via WMMA ----
  v8f sa0 = {}, sa1 = {}, sa2 = {}, sa3 = {};
  {
    const float*  qrow = proj + (size_t)(srow0 + m0 + r16) * PROJ_DIM + qb + koffA;
    const bf16_t* brow = ckb + (size_t)(b * NC + r16) * D_MODEL + hb + koffB;
    #pragma unroll
    for (int k0 = 0; k0 < D_HEAD; k0 += 32) {
      // A-fragment: fp32 q -> bf16 with the softmax scale folded in
      float4 qa = *reinterpret_cast<const float4*>(qrow + k0);
      float4 qc = *reinterpret_cast<const float4*>(qrow + k0 + 4);
      float4 qd = *reinterpret_cast<const float4*>(qrow + k0 + 16);
      float4 qe = *reinterpret_cast<const float4*>(qrow + k0 + 20);
      v16bf af;
      af[0]  = (bf16_t)(qa.x * scale); af[1]  = (bf16_t)(qa.y * scale);
      af[2]  = (bf16_t)(qa.z * scale); af[3]  = (bf16_t)(qa.w * scale);
      af[4]  = (bf16_t)(qc.x * scale); af[5]  = (bf16_t)(qc.y * scale);
      af[6]  = (bf16_t)(qc.z * scale); af[7]  = (bf16_t)(qc.w * scale);
      af[8]  = (bf16_t)(qd.x * scale); af[9]  = (bf16_t)(qd.y * scale);
      af[10] = (bf16_t)(qd.z * scale); af[11] = (bf16_t)(qd.w * scale);
      af[12] = (bf16_t)(qe.x * scale); af[13] = (bf16_t)(qe.y * scale);
      af[14] = (bf16_t)(qe.z * scale); af[15] = (bf16_t)(qe.w * scale);

      v16bf bf0 = load_b_frag(brow + k0);
      v16bf bf1 = load_b_frag(brow + (size_t)16 * D_MODEL + k0);
      v16bf bf2 = load_b_frag(brow + (size_t)32 * D_MODEL + k0);
      v16bf bf3 = load_b_frag(brow + (size_t)48 * D_MODEL + k0);

      sa0 = wmma_bf16(af, bf0, sa0);
      sa1 = wmma_bf16(af, bf1, sa1);
      sa2 = wmma_bf16(af, bf2, sa2);
      sa3 = wmma_bf16(af, bf3, sa3);
    }
  }
  {
    int rb = m0 + (hi ? 8 : 0);
    #pragma unroll
    for (int r = 0; r < 8; ++r) {
      s_sc[(rb + r) * NC + r16]      = sa0[r];
      s_sc[(rb + r) * NC + 16 + r16] = sa1[r];
      s_sc[(rb + r) * NC + 32 + r16] = sa2[r];
      s_sc[(rb + r) * NC + 48 + r16] = sa3[r];
    }
  }
  __syncthreads();

  // ---- phase 2: per-position masked softmax + intra term ----
  if (tid < CHUNK) {
    int s = tid;
    size_t grow = (size_t)(srow0 + s);
    const float* qrow = proj + grow * PROJ_DIM + qb;
    const float* ikr  = intra_k + grow * D_MODEL + hb;
    float lse = 0.f;
    #pragma unroll
    for (int dd = 0; dd < D_HEAD; ++dd) lse += qrow[dd] * ikr[dd];
    lse *= scale;
    float m = lse;
    const float* sr = s_sc + s * NC;
    for (int c = 0; c < cs; ++c) m = fmaxf(m, sr[c]);
    float iw = __expf(lse - m);
    float denom = iw;
    for (int c = 0; c < NC; ++c) {
      float w = (c < cs) ? __expf(sr[c] - m) : 0.f;
      denom += w;
      s_w[s * NC + c] = (bf16_t)w;
    }
    s_iw[s] = iw;
    s_rd[s] = 1.f / denom;
  }
  __syncthreads();

  // ---- phase 3: out = w @ chunk_x via WMMA (A from LDS) ----
  v8f oa0 = {}, oa1 = {}, oa2 = {}, oa3 = {};
  {
    const bf16_t* arow = &s_w[(m0 + r16) * NC + koffA];
    const bf16_t* brow = cxbT + ((size_t)b * D_MODEL + hb + r16) * NC + koffB;
    #pragma unroll
    for (int k0 = 0; k0 < NC; k0 += 32) {
      v16bf af  = load_a_frag_lds(arow + k0);
      v16bf bf0 = load_b_frag(brow + k0);
      v16bf bf1 = load_b_frag(brow + (size_t)16 * NC + k0);
      v16bf bf2 = load_b_frag(brow + (size_t)32 * NC + k0);
      v16bf bf3 = load_b_frag(brow + (size_t)48 * NC + k0);
      oa0 = wmma_bf16(af, bf0, oa0);
      oa1 = wmma_bf16(af, bf1, oa1);
      oa2 = wmma_bf16(af, bf2, oa2);
      oa3 = wmma_bf16(af, bf3, oa3);
    }
  }

  // ---- epilogue ----
  {
    int rb = m0 + (hi ? 8 : 0);
    #pragma unroll
    for (int t = 0; t < 4; ++t) {
      int dd = t * 16 + r16;
      v8f* oa = (t == 0) ? &oa0 : (t == 1) ? &oa1 : (t == 2) ? &oa2 : &oa3;
      #pragma unroll
      for (int r = 0; r < 8; ++r) {
        int s = rb + r;
        size_t grow = (size_t)(srow0 + s);
        float o = (*oa)[r];
        o += s_iw[s] * intra_x[grow * D_MODEL + hb + dd];
        o *= s_rd[s];
        float zv = proj[grow * PROJ_DIM + hb + dd];
        float z = 1.f / (1.f + __expf(-zv));
        zout[grow * D_MODEL + hb + dd] = (bf16_t)(z * o);
      }
    }
  }
}

// ---------------------------------------------------------------------------
// Launch
// ---------------------------------------------------------------------------
extern "C" void kernel_launch(void* const* d_in, const int* in_sizes, int n_in,
                              void* d_out, int out_size, void* d_ws, size_t ws_size,
                              hipStream_t stream) {
  const float* hidden = (const float*)d_in[0];   // (2,4096,1024)
  const float* rmsw   = (const float*)d_in[1];   // (1024,)
  const float* W_in   = (const float*)d_in[2];   // (1024,3328)
  const float* gbias  = (const float*)d_in[3];   // (1024,)
  const float* W_out  = (const float*)d_in[4];   // (1024,1024)
  float* out = (float*)d_out;

  char* ws = (char*)d_ws;
  size_t off = 0;
  bf16_t* hbf   = (bf16_t*)(ws + off); off += (size_t)ROWS * D_MODEL * 2;      // 16 MiB
  bf16_t* wint  = (bf16_t*)(ws + off); off += (size_t)PROJ_DIM * D_MODEL * 2;  // 6.5 MiB
  bf16_t* woutt = (bf16_t*)(ws + off); off += (size_t)D_MODEL * D_MODEL * 2;   // 2 MiB
  float*  proj  = (float*)(ws + off);  off += (size_t)ROWS * PROJ_DIM * 4;     // 104 MiB
  float*  inx   = (float*)(ws + off);  off += (size_t)ROWS * D_MODEL * 4;      // 32 MiB
  float*  ink   = (float*)(ws + off);  off += (size_t)ROWS * D_MODEL * 4;      // 32 MiB
  bf16_t* ckb   = (bf16_t*)(ws + off); off += (size_t)BS * NC * D_MODEL * 2;   // 256 KiB
  bf16_t* cxbT  = (bf16_t*)(ws + off); off += (size_t)BS * D_MODEL * NC * 2;   // 256 KiB
  bf16_t* zo    = (bf16_t*)(ws + off); off += (size_t)ROWS * D_MODEL * 2;      // 16 MiB

  // 1) weight conversion (fp32 KxN -> bf16 NxK)
  {
    size_t tot = (size_t)D_MODEL * PROJ_DIM;
    wt_convert_kernel<<<(int)((tot + 255) / 256), 256, 0, stream>>>(W_in, wint,
                                                                    D_MODEL, PROJ_DIM);
    size_t tot2 = (size_t)D_MODEL * D_MODEL;
    wt_convert_kernel<<<(int)((tot2 + 255) / 256), 256, 0, stream>>>(W_out, woutt,
                                                                     D_MODEL, D_MODEL);
  }
  // 2) rmsnorm
  rmsnorm_kernel<<<ROWS, 256, 0, stream>>>(hidden, rmsw, hbf);
  // 3) GEMM1: proj = h @ W_in
  {
    dim3 grid(ROWS / 128, PROJ_DIM / 64);
    gemm_bf16_kernel<<<grid, 256, 0, stream>>>(hbf, wint, proj, nullptr,
                                               ROWS, PROJ_DIM, D_MODEL);
  }
  // 4) gated chunk scan (+ bf16 chunk summaries in WMMA layouts)
  {
    int total = BS * NC * 2 * D_MODEL;
    scan_kernel<<<(total + 255) / 256, 256, 0, stream>>>(proj, gbias, inx, ink,
                                                         ckb, cxbT);
  }
  // 5) chunk attention entirely on WMMA + z gate -> bf16
  attn_wmma_kernel<<<BS * NUM_HEAD * NC, 128, 0, stream>>>(proj, inx, ink,
                                                           ckb, cxbT, zo);
  // 6) GEMM2 + residual: out = (z*attn) @ W_out + hidden
  {
    dim3 grid(ROWS / 128, D_MODEL / 64);
    gemm_bf16_kernel<<<grid, 256, 0, stream>>>(zo, woutt, out, hidden,
                                               ROWS, D_MODEL, D_MODEL);
  }
}
